// DecomposableAttentionEncoder_48799418417516
// MI455X (gfx1250) — compile-verified, matrix operand order checked by reading, not run on hardware
//
#include <hip/hip_runtime.h>

typedef __bf16 bf16;
typedef __attribute__((ext_vector_type(16))) __bf16 v16bf;
typedef __attribute__((ext_vector_type(8)))  float  v8f;

union FragAB { v16bf v; uint4 u[2]; };

#define TILE_M 128
#define TILE_N 128
#define TILE_K 32

// CDNA5 async global->LDS staging (ASYNCcnt-tracked, no VGPR round-trip).
#define USE_ASYNC_LDS 1
#define LDS_AS __attribute__((address_space(3)))

__device__ __forceinline__ void stage_b128(const bf16* gptr, bf16* lptr) {
#if USE_ASYNC_LDS
  LDS_AS bf16* lp = (LDS_AS bf16*)lptr;   // 32-bit LDS offset in one VGPR
  asm volatile("global_load_async_to_lds_b128 %0, %1, off"
               :: "v"(lp), "v"(gptr) : "memory");
#else
  *(uint4*)lptr = *(const uint4*)gptr;
#endif
}

__device__ __forceinline__ void stage_fence() {
#if USE_ASYNC_LDS
  asm volatile("s_wait_asynccnt 0" ::: "memory");
#endif
}

// ---------------------------------------------------------------------------
// elementwise f32 -> bf16
// ---------------------------------------------------------------------------
__global__ __launch_bounds__(256)
void f32_to_bf16_kernel(const float* __restrict__ in, bf16* __restrict__ out, long n) {
  long i = (long)blockIdx.x * blockDim.x + threadIdx.x;
  long stride = (long)gridDim.x * blockDim.x;
  for (; i < n; i += stride) out[i] = (bf16)in[i];
}

// ---------------------------------------------------------------------------
// Generic batched WMMA GEMM:  C = act( A(bf16) @ B(bf16[^T]) [+ Cacc] [+ bias] )
// Block = 256 threads = 8 waves (wave32). Tile 128x128, K-step 32, K%64==0.
// Waves arranged 2 (M) x 4 (N); each wave owns 4x2 subtiles of 16x16
// -> 8 v_wmma_f32_16x16x32_bf16 per K-step per wave.
//
// Double-buffered LDS + GLOBAL_LOAD_ASYNC_TO_LDS_B128: tile k+1 streams in
// (ASYNCcnt) while tile k runs through the WMMA pipe; one workgroup barrier
// per K-step. Template-specialized on TRANSB / GUARDM so the hot loop carries
// no runtime flag checks and staging pointers are simple induction variables.
//
// LDS layout: As[row][k] row-major; Bs[col][k] K-contiguous per output column
// so A and B fragments each load as 2x ds_load_b128 per lane, matching the
// ISA 16-bit operand layout:
//   lanes 0-15  : K = 0..7  (VGPR0-3), K = 16..23 (VGPR4-7)
//   lanes 16-31 : K = 8..15 (VGPR0-3), K = 24..31 (VGPR4-7)
// ---------------------------------------------------------------------------
template <int TRANSB, int GUARDM>
__global__ __launch_bounds__(256)
void wmma_gemm_bf16(const bf16* __restrict__ A, long sA, int lda,
                    const bf16* __restrict__ Bm, long sB, int ldb,
                    const float* __restrict__ bias,
                    const float* __restrict__ Cacc, long sCa, int ldca,
                    float* __restrict__ Cf, long sCf, int ldcf,
                    bf16* __restrict__ Cb, long sCb, int ldcb,
                    int Mrows, int K, int relu) {
  __shared__ bf16 As[2][TILE_M][TILE_K];
  __shared__ bf16 Bs[2][TILE_N][TILE_K];

  const int tid  = threadIdx.x;
  const int lane = tid & 31;
  const int wave = tid >> 5;
  const int wm   = wave >> 2;   // 0..1 -> 64 rows each
  const int wn   = wave & 3;    // 0..3 -> 32 cols each
  const int lr   = lane & 15;
  const int lh   = lane >> 4;

  const int m0 = blockIdx.y * TILE_M;
  const int n0 = blockIdx.x * TILE_N;
  const int z  = blockIdx.z;

  A  += (long)z * sA;
  Bm += (long)z * sB;
  const float* CaccZ = Cacc ? Cacc + (long)z * sCa : (const float*)0;
  float*       CfZ   = Cf   ? Cf   + (long)z * sCf : (float*)0;
  bf16*        CbZ   = Cb   ? Cb   + (long)z * sCb : (bf16*)0;

  // ---- per-thread staging geometry (chunk pair = row / row+64, same ko) ----
  const int ld_row = tid >> 2;          // 0..63
  const int ld_ko  = (tid & 3) * 8;     // 0/8/16/24
  const bf16* aP0 = A + (long)(m0 + ld_row) * lda + ld_ko;
  const bf16* aP1 = aP0 + (long)64 * lda;
  bool aV0 = true, aV1 = true;
  if (GUARDM) {
    aV0 = (m0 + ld_row)      < Mrows;
    aV1 = (m0 + ld_row + 64) < Mrows;
  }
  // TRANSB staging: columns of C = rows of B, K-contiguous.
  const bf16* bP0 = Bm + (long)(n0 + ld_row) * ldb + ld_ko;
  const bf16* bP1 = bP0 + (long)64 * ldb;
  // non-trans staging: rows k / k+16, 8 contiguous N elems.
  const int ld_k  = tid >> 4;           // 0..15
  const int ld_no = (tid & 15) * 8;     // 0..120
  const bf16* bPn = Bm + (long)ld_k * ldb + n0 + ld_no;

  auto stageTile = [&](int buf, int k0) {
    if (GUARDM) {
      uint4 u0 = {0u,0u,0u,0u}, u1 = {0u,0u,0u,0u};
      if (aV0) u0 = *(const uint4*)(aP0 + k0);
      if (aV1) u1 = *(const uint4*)(aP1 + k0);
      *(uint4*)&As[buf][ld_row][ld_ko]      = u0;
      *(uint4*)&As[buf][ld_row + 64][ld_ko] = u1;
    } else {
      stage_b128(aP0 + k0, &As[buf][ld_row][ld_ko]);
      stage_b128(aP1 + k0, &As[buf][ld_row + 64][ld_ko]);
    }
    if (TRANSB) {
      stage_b128(bP0 + k0, &Bs[buf][ld_row][ld_ko]);
      stage_b128(bP1 + k0, &Bs[buf][ld_row + 64][ld_ko]);
    } else {
      union { uint4 u; bf16 h[8]; } t0, t1;
      t0.u = *(const uint4*)(bPn + (long)k0 * ldb);
      t1.u = *(const uint4*)(bPn + (long)(k0 + 16) * ldb);
#pragma unroll
      for (int j = 0; j < 8; ++j) {
        Bs[buf][ld_no + j][ld_k]      = t0.h[j];
        Bs[buf][ld_no + j][ld_k + 16] = t1.h[j];
      }
    }
  };

  v8f acc[4][2] = {};

  auto computeTile = [&](int buf) {
    FragAB fa[4], fb[2];
#pragma unroll
    for (int im = 0; im < 4; ++im) {
      int r = wm * 64 + im * 16 + lr;
      fa[im].u[0] = *(const uint4*)&As[buf][r][lh * 8];
      fa[im].u[1] = *(const uint4*)&As[buf][r][16 + lh * 8];
    }
#pragma unroll
    for (int in = 0; in < 2; ++in) {
      int cc = wn * 32 + in * 16 + lr;
      fb[in].u[0] = *(const uint4*)&Bs[buf][cc][lh * 8];
      fb[in].u[1] = *(const uint4*)&Bs[buf][cc][16 + lh * 8];
    }
#pragma unroll
    for (int im = 0; im < 4; ++im)
#pragma unroll
      for (int in = 0; in < 2; ++in)
        acc[im][in] = __builtin_amdgcn_wmma_f32_16x16x32_bf16(
            false, fa[im].v, false, fb[in].v, (short)0, acc[im][in], false, false);
  };

  // ---- software-pipelined main loop (K % 64 == 0) ----
  stageTile(0, 0);
  stage_fence();
  __syncthreads();
  for (int k0 = 0; k0 < K; k0 += 2 * TILE_K) {
    stageTile(1, k0 + TILE_K);     // streams while buf0 computes
    computeTile(0);
    stage_fence();
    __syncthreads();
    if (k0 + 2 * TILE_K < K) stageTile(0, k0 + 2 * TILE_K);
    computeTile(1);
    stage_fence();
    __syncthreads();
  }

  // ---- epilogue: [+Cacc] +bias, ReLU, dual-format store with own ldc ----
#pragma unroll
  for (int im = 0; im < 4; ++im) {
#pragma unroll
    for (int in = 0; in < 2; ++in) {
      int col = n0 + wn * 32 + in * 16 + lr;
      float bv = bias ? bias[col] : 0.0f;
#pragma unroll
      for (int i = 0; i < 8; ++i) {
        int r = m0 + wm * 64 + im * 16 + lh * 8 + i;
        if (!GUARDM || r < Mrows) {
          float v = acc[im][in][i];
          if (CaccZ) v += CaccZ[(long)r * ldca + col];
          v += bv;
          if (relu) v = fmaxf(v, 0.0f);
          if (CfZ) CfZ[(long)r * ldcf + col] = v;
          if (CbZ) CbZ[(long)r * ldcb + col] = (bf16)v;
        }
      }
    }
  }
}

// ---------------------------------------------------------------------------
// Row softmax over 512 columns, optional relative-distance bias
// (d = clip(col-row,-11,11)+11), optional transposed read (for softmax(X^T)).
// One block per (batch z, row). Output bf16 (feeds next WMMA GEMM).
// Masks in this problem are all-true, so plain softmax.
// ---------------------------------------------------------------------------
__global__ __launch_bounds__(256)
void softmax512_kernel(const float* __restrict__ X, long sX, int ldx, int transIn,
                       const float* __restrict__ dist, int useDist,
                       bf16* __restrict__ O, long sO, int ldo) {
  const int tid = threadIdx.x;
  const int row = blockIdx.x;
  const int z   = blockIdx.z;
  X += (long)z * sX;
  O += (long)z * sO;
  __shared__ float red[256];

  float x[2];
#pragma unroll
  for (int i = 0; i < 2; ++i) {
    int col = tid + i * 256;
    float v = transIn ? X[(long)col * ldx + row] : X[(long)row * ldx + col];
    if (useDist) {
      int d = col - row;
      d = d < -11 ? -11 : (d > 11 ? 11 : d);
      v += dist[d + 11];
    }
    x[i] = v;
  }
  float m = fmaxf(x[0], x[1]);
  red[tid] = m; __syncthreads();
  for (int s = 128; s > 0; s >>= 1) {
    if (tid < s) red[tid] = fmaxf(red[tid], red[tid + s]);
    __syncthreads();
  }
  m = red[0]; __syncthreads();
  float e0 = __expf(x[0] - m), e1 = __expf(x[1] - m);
  red[tid] = e0 + e1; __syncthreads();
  for (int s = 128; s > 0; s >>= 1) {
    if (tid < s) red[tid] += red[tid + s];
    __syncthreads();
  }
  float inv = 1.0f / red[0];
  O[(long)row * ldo + (tid)]       = (bf16)(e0 * inv);
  O[(long)row * ldo + (tid + 256)] = (bf16)(e1 * inv);
}

// ---------------------------------------------------------------------------
// pooled[b][h] = sum_m X[b][m][h]   (mask all-true). out pointer pre-offset.
// ---------------------------------------------------------------------------
__global__ __launch_bounds__(256)
void colsum_kernel(const float* __restrict__ X, float* __restrict__ out,
                   int Mrows, int ncols, int ldout) {
  int b = blockIdx.y;
  int h = blockIdx.x * 256 + threadIdx.x;
  const float* p = X + (long)b * Mrows * ncols + h;
  float s = 0.0f;
  for (int m = 0; m < Mrows; ++m) s += p[(long)m * ncols];
  out[(long)b * ldout + h] = s;
}

// ---------------------------------------------------------------------------
extern "C" void kernel_launch(void* const* d_in, const int* in_sizes, int n_in,
                              void* d_out, int out_size, void* d_ws, size_t ws_size,
                              hipStream_t stream) {
  (void)in_sizes; (void)n_in; (void)out_size; (void)ws_size;

  const float* prem = (const float*)d_in[0];
  const float* hypo = (const float*)d_in[1];
  // d_in[2], d_in[3]: masks (all-true) — unused
  const float* Wpx = (const float*)d_in[4];  const float* bpx = (const float*)d_in[5];
  const float* Wpy = (const float*)d_in[6];  const float* bpy = (const float*)d_in[7];
  const float* dist= (const float*)d_in[8];
  const float* Ws1 = (const float*)d_in[9];  const float* bs1 = (const float*)d_in[10];
  const float* Ws2 = (const float*)d_in[11]; const float* bs2 = (const float*)d_in[12];
  const float* Wa1 = (const float*)d_in[13]; const float* ba1 = (const float*)d_in[14];
  const float* Wa2 = (const float*)d_in[15]; const float* ba2 = (const float*)d_in[16];
  const float* Wc1 = (const float*)d_in[17]; const float* bc1 = (const float*)d_in[18];
  const float* Wc2 = (const float*)d_in[19]; const float* bc2 = (const float*)d_in[20];
  const float* Wg1 = (const float*)d_in[21]; const float* bg1 = (const float*)d_in[22];
  const float* Wg2 = (const float*)d_in[23]; const float* bg2 = (const float*)d_in[24];

  const int  Bb = 32, M = 512, S = 512;
  const int  Mtot = Bb * M;                 // 16384
  const long BMS  = (long)Bb * M * S;       // 8,388,608

  // ---- bump allocator over d_ws ----
  char* p = (char*)d_ws;
  auto alloc = [&](size_t bytes) -> char* {
    char* r = p; p += (bytes + 255) & ~(size_t)255; return r;
  };

  bf16* Wpx_b = (bf16*)alloc((size_t)512 * 512 * 2);
  bf16* Wpy_b = (bf16*)alloc((size_t)512 * 512 * 2);
  bf16* Ws1_b = (bf16*)alloc((size_t)512 * 512 * 2);
  bf16* Ws2_b = (bf16*)alloc((size_t)512 * 512 * 2);
  bf16* Wa1_b = (bf16*)alloc((size_t)1024 * 512 * 2);
  bf16* Wa2_b = (bf16*)alloc((size_t)512 * 512 * 2);
  bf16* Wc1_b = (bf16*)alloc((size_t)2048 * 512 * 2);
  bf16* Wc2_b = (bf16*)alloc((size_t)512 * 512 * 2);
  bf16* Wg1_b = (bf16*)alloc((size_t)1024 * 512 * 2);
  bf16* Wg2_b = (bf16*)alloc((size_t)512 * 512 * 2);

  bf16*  pb16    = (bf16*)alloc((size_t)BMS * 2);
  bf16*  hb16    = (bf16*)alloc((size_t)BMS * 2);
  bf16*  prem2_b = (bf16*)alloc((size_t)BMS * 2 * 2);   // [B,M,2S]
  bf16*  hypo2_b = (bf16*)alloc((size_t)BMS * 2 * 2);   // [B,N,2S]
  bf16*  t1_b    = (bf16*)alloc((size_t)BMS * 2);       // hidden scratch
  bf16*  fx_b    = (bf16*)alloc((size_t)BMS * 2);       // fp/fh/gp
  bf16*  gh_b    = (bf16*)alloc((size_t)BMS * 2);       // gh
  float* scr_f   = (float*)alloc((size_t)BMS * 4);      // scores / sim
  bf16*  prob_b  = (bf16*)alloc((size_t)BMS * 2);       // attention probs
  bf16*  ah_b    = (bf16*)alloc((size_t)BMS * 2 * 2);   // attended_hypo [B,M,2S]
  bf16*  ap_b    = (bf16*)alloc((size_t)BMS * 2 * 2);   // attended_prem [B,N,2S]
  float* acc_f   = (float*)alloc((size_t)BMS * 4);      // split-K accumulator
  float* cmp_f   = (float*)alloc((size_t)BMS * 4);      // compare output
  float* pooled_f= (float*)alloc((size_t)Bb * 1024 * 4);
  bf16*  pooled_b= (bf16*)alloc((size_t)Bb * 1024 * 2);
  bf16*  tg_b    = (bf16*)alloc((size_t)Bb * 512 * 2);

  auto cvt = [&](const float* src, bf16* dst, long n) {
    int grid = (int)((n + 255) / 256); if (grid > 4096) grid = 4096;
    f32_to_bf16_kernel<<<grid, 256, 0, stream>>>(src, dst, n);
  };
  auto gemm = [&](const bf16* A, long sA, int lda,
                  const bf16* Bm, long sB, int ldb, int transB,
                  const float* bias,
                  const float* Cacc, int ldca, long sCa,
                  float* Cf, int ldcf, long sCf,
                  bf16* Cb, int ldcb, long sCb,
                  int Mrows, int Ncols, int K, int relu, int batches) {
    dim3 grid(Ncols / TILE_N, (Mrows + TILE_M - 1) / TILE_M, batches);
    bool g = (Mrows % TILE_M) != 0;
    if (transB) {
      if (g) wmma_gemm_bf16<1,1><<<grid, 256, 0, stream>>>(A, sA, lda, Bm, sB, ldb,
                 bias, Cacc, sCa, ldca, Cf, sCf, ldcf, Cb, sCb, ldcb, Mrows, K, relu);
      else   wmma_gemm_bf16<1,0><<<grid, 256, 0, stream>>>(A, sA, lda, Bm, sB, ldb,
                 bias, Cacc, sCa, ldca, Cf, sCf, ldcf, Cb, sCb, ldcb, Mrows, K, relu);
    } else {
      if (g) wmma_gemm_bf16<0,1><<<grid, 256, 0, stream>>>(A, sA, lda, Bm, sB, ldb,
                 bias, Cacc, sCa, ldca, Cf, sCf, ldcf, Cb, sCb, ldcb, Mrows, K, relu);
      else   wmma_gemm_bf16<0,0><<<grid, 256, 0, stream>>>(A, sA, lda, Bm, sB, ldb,
                 bias, Cacc, sCa, ldca, Cf, sCf, ldcf, Cb, sCb, ldcb, Mrows, K, relu);
    }
  };
  auto softmax = [&](const float* X, int transIn, int useDist, bf16* O) {
    softmax512_kernel<<<dim3(512, 1, Bb), 256, 0, stream>>>(
        X, 512L * 512, 512, transIn, dist, useDist, O, 512L * 512, 512);
  };

  // ---- phase 0: convert inputs + weights to bf16 ----
  cvt(prem, pb16, BMS);
  cvt(hypo, hb16, BMS);
  cvt(Wpx, Wpx_b, 512L * 512);  cvt(Wpy, Wpy_b, 512L * 512);
  cvt(Ws1, Ws1_b, 512L * 512);  cvt(Ws2, Ws2_b, 512L * 512);
  cvt(Wa1, Wa1_b, 1024L * 512); cvt(Wa2, Wa2_b, 512L * 512);
  cvt(Wc1, Wc1_b, 2048L * 512); cvt(Wc2, Wc2_b, 512L * 512);
  cvt(Wg1, Wg1_b, 1024L * 512); cvt(Wg2, Wg2_b, 512L * 512);

  // ---- phase 1: projections -> columns [0,512) of *2 buffers (ldc=1024) ----
  gemm(pb16, 0, 512, Wpy_b, 0, 512, 0, bpy, 0, 0, 0, 0, 0, 0,
       prem2_b, 1024, 0, Mtot, 512, 512, 0, 1);
  gemm(hb16, 0, 512, Wpx_b, 0, 512, 0, bpx, 0, 0, 0, 0, 0, 0,
       hypo2_b, 1024, 0, Mtot, 512, 512, 0, 1);

  // ---- phase 2+3: self attention (prem side, then hypo side) ----
  for (int side = 0; side < 2; ++side) {
    bf16* x2 = side == 0 ? prem2_b : hypo2_b;
    gemm(x2, 0, 1024, Ws1_b, 0, 512, 0, bs1, 0, 0, 0, 0, 0, 0,
         t1_b, 512, 0, Mtot, 512, 512, 1, 1);
    gemm(t1_b, 0, 512, Ws2_b, 0, 512, 0, bs2, 0, 0, 0, 0, 0, 0,
         fx_b, 512, 0, Mtot, 512, 512, 1, 1);
    // scores = f @ f^T + rel_bias  (batched, both operands per-batch)
    gemm(fx_b, 512L * 512, 512, fx_b, 512L * 512, 512, 1, 0, 0, 0, 0,
         scr_f, 512, 512L * 512, 0, 0, 0, 512, 512, 512, 0, Bb);
    softmax(scr_f, 0, 1, prob_b);
    // ctx = probs @ x_p  -> columns [512,1024) of x2 (concat-in-place)
    gemm(prob_b, 512L * 512, 512, x2, 1024L * 512, 1024, 0, 0, 0, 0, 0,
         0, 0, 0, x2 + 512, 1024, 1024L * 512, 512, 512, 512, 0, Bb);
  }

  // ---- phase 4: cross attention ----
  gemm(prem2_b, 0, 1024, Wa1_b, 0, 512, 0, ba1, 0, 0, 0, 0, 0, 0,
       t1_b, 512, 0, Mtot, 512, 1024, 1, 1);
  gemm(t1_b, 0, 512, Wa2_b, 0, 512, 0, ba2, 0, 0, 0, 0, 0, 0,
       fx_b, 512, 0, Mtot, 512, 512, 1, 1);                       // gp
  gemm(hypo2_b, 0, 1024, Wa1_b, 0, 512, 0, ba1, 0, 0, 0, 0, 0, 0,
       t1_b, 512, 0, Mtot, 512, 1024, 1, 1);
  gemm(t1_b, 0, 512, Wa2_b, 0, 512, 0, ba2, 0, 0, 0, 0, 0, 0,
       gh_b, 512, 0, Mtot, 512, 512, 1, 1);                       // gh
  // sim = gp @ gh^T
  gemm(fx_b, 512L * 512, 512, gh_b, 512L * 512, 512, 1, 0, 0, 0, 0,
       scr_f, 512, 512L * 512, 0, 0, 0, 512, 512, 512, 0, Bb);
  softmax(scr_f, 0, 0, prob_b);                                   // p2h
  gemm(prob_b, 512L * 512, 512, hypo2_b, 1024L * 512, 1024, 0, 0, 0, 0, 0,
       0, 0, 0, ah_b, 1024, 1024L * 512, 512, 1024, 512, 0, Bb);  // attended_hypo
  softmax(scr_f, 1, 0, prob_b);                                   // h2p
  gemm(prob_b, 512L * 512, 512, prem2_b, 1024L * 512, 1024, 0, 0, 0, 0, 0,
       0, 0, 0, ap_b, 1024, 1024L * 512, 512, 1024, 512, 0, Bb);  // attended_prem

  // ---- phase 5: compare + pool (split-K over the 4S concat) ----
  for (int side = 0; side < 2; ++side) {
    bf16* x2 = side == 0 ? prem2_b : hypo2_b;
    bf16* at = side == 0 ? ah_b    : ap_b;
    gemm(x2, 0, 1024, Wc1_b, 0, 512, 0, 0, 0, 0, 0,
         acc_f, 512, 0, 0, 0, 0, Mtot, 512, 1024, 0, 1);
    gemm(at, 0, 1024, Wc1_b + (size_t)1024 * 512, 0, 512, 0, bc1,
         acc_f, 512, 0, 0, 0, 0, t1_b, 512, 0, Mtot, 512, 1024, 1, 1);
    gemm(t1_b, 0, 512, Wc2_b, 0, 512, 0, bc2, 0, 0, 0,
         cmp_f, 512, 0, 0, 0, 0, Mtot, 512, 512, 1, 1);
    colsum_kernel<<<dim3(2, Bb), 256, 0, stream>>>(cmp_f, pooled_f + side * 512,
                                                   512, 512, 1024);
  }

  // ---- phase 6: aggregate MLP (M=32, row guards active) ----
  cvt(pooled_f, pooled_b, (long)Bb * 1024);
  gemm(pooled_b, 0, 1024, Wg1_b, 0, 512, 0, bg1, 0, 0, 0, 0, 0, 0,
       tg_b, 512, 0, Bb, 512, 1024, 1, 1);
  gemm(tg_b, 0, 512, Wg2_b, 0, 512, 0, bg2, 0, 0, 0,
       (float*)d_out, 512, 0, 0, 0, 0, Bb, 512, 512, 1, 1);
}
